// MinLSTM_19585050870363
// MI455X (gfx1250) — matrix operand order repaired and checked by reading, not compiled, not run on hardware
//
#include <hip/hip_runtime.h>
#include <hip/hip_bf16.h>

typedef __attribute__((ext_vector_type(16))) _Float16 v16h;
typedef __attribute__((ext_vector_type(8)))  float    v8f;
typedef __attribute__((ext_vector_type(8)))  unsigned int v8u;

#define S_DIM 10
#define T_LEN 2048
#define CT 32                      // timesteps staged per LDS chunk
#define ROWS_PER_BLOCK 128         // 8 waves * 16 batch columns
#define XBUF_STRIDE (ROWS_PER_BLOCK + 1)  // pad: t-stride 129 -> conflict-free

// Fast sigmoid via gfx1250 hardware v_tanh_f32.  The 0.5 input scale is
// pre-folded into the gate weight matrices (GATE_PRESCALE), so at runtime
// sigma(z) costs just v_tanh + v_fma per element.
#if __has_builtin(__builtin_amdgcn_tanhf)
#define FAST_TANH(x) __builtin_amdgcn_tanhf(x)
#define HAVE_HW_TANH 1
#elif __has_builtin(__builtin_amdgcn_tanh_f32)
#define FAST_TANH(x) __builtin_amdgcn_tanh_f32(x)
#define HAVE_HW_TANH 1
#else
#define HAVE_HW_TANH 0
#endif

#if HAVE_HW_TANH
#define GATE_PRESCALE 0.5f
#define SIGMOID_PRE(zh) (__builtin_fmaf(0.5f, FAST_TANH(zh), 0.5f))  // zh = z/2
#else
#define GATE_PRESCALE 1.0f
#define SIGMOID_PRE(z)  (__builtin_amdgcn_rcpf(1.0f + __expf(-(z))))
#endif

// One v_cvt_pk_rtz_f16_f32: two f32 -> packed half2 in a dword.
static __device__ __forceinline__ unsigned int pk2(float a, float b) {
    auto p = __builtin_amdgcn_cvt_pkrtz(a, b);   // __fp16 ext_vector(2)
    return __builtin_bit_cast(unsigned int, p);
}

// Cross-half swap (lane i <-> lane i^16) as a single VALU op.
static __device__ __forceinline__ unsigned int xhalf(unsigned int v) {
#if __has_builtin(__builtin_amdgcn_permlanex16)
    return __builtin_amdgcn_permlanex16(v, v, 0x76543210u, 0xfedcba98u, false, false);
#else
    return (unsigned int)__shfl_xor((int)v, 16, 32);
#endif
}

__global__ __launch_bounds__(256) void minlstm_wmma_kernel(
    const float* __restrict__ X,
    const float* __restrict__ Wui, const float* __restrict__ bui,
    const float* __restrict__ Wus, const float* __restrict__ bus,
    const float* __restrict__ Wfi, const float* __restrict__ bfi,
    const float* __restrict__ Wfs, const float* __restrict__ bfs,
    const float* __restrict__ Wni, const float* __restrict__ bni,
    const float* __restrict__ Wns, const float* __restrict__ bns,
    const float* __restrict__ Wr,  const float* __restrict__ br,
    float* __restrict__ out)
{
    __shared__ float xbuf[CT * XBUF_STRIDE];

    const int  tid  = threadIdx.x;
    const int  lane = tid & 31;
    const int  wave = tid >> 5;
    const int  col  = lane & 15;     // batch column within the 16-wide tile
    const bool lo   = lane < 16;
    const int  m    = col;           // A-matrix output row this lane holds

    const int block_row0 = blockIdx.x * ROWS_PER_BLOCK;
    const int wave_row0  = block_row0 + wave * 16;

    // ---- Build constant A fragments: W' = [W(SxS) | w_in | bias | 0...] as
    //      16x32 f16.  This code's A k order (B matches): lanes 0-15 hold M=lane with
    //      K = {0..15}; lanes 16-31 hold M=lane-16 with K={16..31}.
    //      Gate matrices (u,f) are pre-scaled by 0.5 for the tanh sigmoid.
    v16h a_u, a_f, a_n;
    {
        const int kbase = lo ? 0 : 16;
        #pragma unroll
        for (int i = 0; i < 16; ++i) {
            const int K = i + kbase;
            float vu = 0.f, vf = 0.f, vn = 0.f;
            if (m < S_DIM) {
                if (K < S_DIM) {
                    vu = Wus[m * S_DIM + K];
                    vf = Wfs[m * S_DIM + K];
                    vn = Wns[m * S_DIM + K];
                } else if (K == S_DIM) {          // scalar-input weight column
                    vu = Wui[m]; vf = Wfi[m]; vn = Wni[m];
                } else if (K == S_DIM + 1) {      // fused bias column (x "1" row)
                    vu = bui[m] + bus[m];
                    vf = bfi[m] + bfs[m];
                    vn = bni[m] + bns[m];
                }
            }
            a_u[i] = (_Float16)(vu * GATE_PRESCALE);
            a_f[i] = (_Float16)(vf * GATE_PRESCALE);
            a_n[i] = (_Float16)vn;
        }
    }

    // Readout weights for the states this lane's D registers hold:
    // D VGPR r <-> state (r + (lo?0:8)) at batch column `col`.
    float wr_l[8];
    #pragma unroll
    for (int r = 0; r < 8; ++r) {
        const int s = r + (lo ? 0 : 8);
        wr_l[r] = (s < S_DIM) ? Wr[s] : 0.f;
    }

    const unsigned int msk = lo ? 0xFFFFFFFFu : 0u;  // zero B rows K>=16

    v8f h = {};                       // state tile, f32 C/D layout
    const v8f cz = {};                // zero accumulator (bias lives in A)

    const int nchunks = T_LEN / CT;
    for (int c = 0; c < nchunks; ++c) {
        // ---- stage X[block rows, c*CT .. c*CT+CT) into LDS, transposed ----
        __syncthreads();
        #pragma unroll
        for (int j = 0; j < ROWS_PER_BLOCK / 8; ++j) {
            const int row = wave + j * 8;             // whole wave reads one row:
            xbuf[lane * XBUF_STRIDE + row] =          // 128B contiguous, coalesced
                X[(long)(block_row0 + row) * T_LEN + c * CT + lane];
        }
        __syncthreads();

        // Prefetch next chunk into cache while we compute this one
        // (global_prefetch_b8; no counters, fire-and-forget).
        if (c + 1 < nchunks) {
            #pragma unroll
            for (int j = 0; j < ROWS_PER_BLOCK / 8; ++j) {
                const int row = wave + j * 8;
                __builtin_prefetch(
                    &X[(long)(block_row0 + row) * T_LEN + (c + 1) * CT + lane], 0, 1);
            }
        }

        if (c == 0) {
            // h0: state 0 = X[:,0], everything else zero
            const float x0 = xbuf[0 * XBUF_STRIDE + wave * 16 + col];
            h[0] = lo ? x0 : 0.f;
        }

        for (int t = (c == 0) ? 1 : 0; t < CT; ++t) {
            const float x = xbuf[t * XBUF_STRIDE + wave * 16 + col];

            // ---- B fragment = H'^T (32x16 f16): lanes 0-15 hold K=0..15,
            //      lanes 16-31 hold K=16..31 (zeroed padding rows).
            // p01 on hi lanes is packed(state8, state9) for this column, so a
            // single v_permlanex16 of the packed dword yields row pair K=8,9.
            const unsigned int p01 = pk2(h[0], h[1]);
            v8u bu;
            bu[0] = p01 & msk;
            bu[1] = pk2(h[2], h[3]) & msk;
            bu[2] = pk2(h[4], h[5]) & msk;
            bu[3] = pk2(h[6], h[7]) & msk;
            bu[4] = xhalf(p01) & msk;        // states 8,9 from the hi half
            bu[5] = pk2(x, 1.0f) & msk;      // input row K=10, bias row K=11
            bu[6] = 0u;
            bu[7] = 0u;
            const v16h b = __builtin_bit_cast(v16h, bu);

            // ---- three gate pre-activations in one WMMA each ----
            v8f zu = __builtin_amdgcn_wmma_f32_16x16x32_f16(
                false, a_u, false, b, (short)0, cz, false, false);
            v8f zf = __builtin_amdgcn_wmma_f32_16x16x32_f16(
                false, a_f, false, b, (short)0, cz, false, false);
            v8f zn = __builtin_amdgcn_wmma_f32_16x16x32_f16(
                false, a_n, false, b, (short)0, cz, false, false);

            // ---- gated update: h = sigmoid(zu)*zn + sigmoid(zf)*h ----
            #pragma unroll
            for (int r = 0; r < 8; ++r) {
                const float gi = SIGMOID_PRE(zu[r]);
                const float gf = SIGMOID_PRE(zf[r]);
                h[r] = gi * zn[r] + gf * h[r];
            }
        }
    }

    // ---- readout: out[b] = sum_s h[b,s] * Wr[s] + br ----
    float acc = 0.f;
    #pragma unroll
    for (int r = 0; r < 8; ++r) acc += h[r] * wr_l[r];
    acc += __shfl_xor(acc, 16, 32);          // combine states 0-7 with 8-15
    if (lo) out[wave_row0 + col] = acc + br[0];
}

extern "C" void kernel_launch(void* const* d_in, const int* in_sizes, int n_in,
                              void* d_out, int out_size, void* d_ws, size_t ws_size,
                              hipStream_t stream) {
    (void)in_sizes; (void)n_in; (void)out_size; (void)d_ws; (void)ws_size;
    const float* X   = (const float*)d_in[0];
    const float* Wui = (const float*)d_in[1];
    const float* bui = (const float*)d_in[2];
    const float* Wus = (const float*)d_in[3];
    const float* bus = (const float*)d_in[4];
    const float* Wfi = (const float*)d_in[5];
    const float* bfi = (const float*)d_in[6];
    const float* Wfs = (const float*)d_in[7];
    const float* bfs = (const float*)d_in[8];
    const float* Wni = (const float*)d_in[9];
    const float* bni = (const float*)d_in[10];
    const float* Wns = (const float*)d_in[11];
    const float* bns = (const float*)d_in[12];
    const float* Wr  = (const float*)d_in[13];
    const float* br  = (const float*)d_in[14];
    float* out = (float*)d_out;

    dim3 grid(16384 / ROWS_PER_BLOCK);   // 128 blocks
    dim3 block(256);                     // 8 waves -> 128 batch rows per block
    hipLaunchKernelGGL(minlstm_wmma_kernel, grid, block, 0, stream,
                       X, Wui, bui, Wus, bus, Wfi, bfi, Wfs, bfs,
                       Wni, bni, Wns, bns, Wr, br, out);
}
